// MLP_90125593739290
// MI455X (gfx1250) — compile-verified, matrix operand order checked by reading, not run on hardware
//
#include <hip/hip_runtime.h>
#include <hip/hip_bf16.h>

#define KNN 20

typedef __attribute__((ext_vector_type(16))) __bf16 v16bf;
typedef __attribute__((ext_vector_type(8)))  float  v8f;

__device__ __forceinline__ unsigned short f2bf(float f) {
  unsigned int u = __float_as_uint(f);
  u += 0x7fffu + ((u >> 16) & 1u);   // round-to-nearest-even
  return (unsigned short)(u >> 16);
}

// ---------------------------------------------------------------------------
// WMMA GEMM:  Out[M,P] = W[M,CK] * Bmat[CK,P]   (bf16 inputs, f32 accumulate)
// W pre-packed in A-fragment order. B tile built cooperatively in LDS, in
// WMMA fragment order, once per block; 8 waves * 2 M-tiles each share it, so
// each im2col column is gathered once per 256 output rows.
// MODE 0: conv2d 3x3 over (OH,OW), input [Cin][Bb][IH*IW], pad 0/1
// MODE 1: conv1d k=3 over OH(=OL), input [Cin][Bb][IH(=IL)], pad 0/1
// MODE 2: k=1 / linear: input [CK][Bb][IH(=S)], OH=S
// Epilogue: ep==1 -> y = s*v+b, leaky_relu(0.2); ep==2 -> y = v + bias[m]
// ---------------------------------------------------------------------------
template<int MODE>
__global__ __launch_bounds__(256)
void k_gemm(const uint4* __restrict__ pA, const float* __restrict__ in,
            float* __restrict__ out, const float* __restrict__ scale,
            const float* __restrict__ bias, int M, int CK, int P, int ep,
            int Bb, int OH, int OW, int IH, int IW, int pad)
{
  __shared__ unsigned int lds[2][256];   // [buf][lane*8 + dword] = B fragment

  const int tid  = threadIdx.x;
  const int lane = tid & 31;
  const int wv   = tid >> 5;
  const int Mt  = (M + 15) >> 4;
  const int MtG = (Mt + 15) >> 4;        // 16 M-tiles per block (2 per wave)
  const int mtg = blockIdx.x % MtG;
  const int pt  = blockIdx.x / MtG;
  const int mt0 = mtg * 16 + wv * 2;
  const int mt1 = mt0 + 1;
  const bool mv0 = mt0 < Mt;
  const bool mv1 = mt1 < Mt;

  // ---- staging role: this thread fills LDS dword 'tid' = (slane, si) -------
  const int si     = tid & 7;            // dword within lane's fragment
  const int slane  = tid >> 3;           // fragment lane this thread feeds
  const int e0     = 2 * si;
  const int skoff  = ((slane >= 16) ? 8 : 0) + e0 + ((e0 >= 8) ? 8 : 0);
  int pcolS = pt * 16 + (slane & 15);
  int pcS   = (pcolS < P) ? pcolS : (P - 1);
  int bS = 0, oyS = 0, oxS = 0;
  if (MODE == 0) {
    int OS = OH * OW; bS = pcS / OS; int s = pcS - bS * OS; oyS = s / OW; oxS = s - oyS * OW;
  } else {
    bS = pcS / OH; oyS = pcS - bS * OH;
  }
  auto fetchB = [&](int k) -> float {
    if (k >= CK) return 0.f;
    if (MODE == 0) {
      int ci = k / 9; int r = k - ci * 9; int dy = r / 3; int dx = r - dy * 3;
      int iy = oyS + dy - pad; int ix = oxS + dx - pad;
      if ((unsigned)iy < (unsigned)IH && (unsigned)ix < (unsigned)IW)
        return in[((size_t)ci * Bb + bS) * ((size_t)IH * IW) + (size_t)iy * IW + ix];
      return 0.f;
    } else if (MODE == 1) {
      int ci = k / 3; int dx = k - ci * 3;
      int ix = oyS + dx - pad;
      if ((unsigned)ix < (unsigned)IH)
        return in[((size_t)ci * Bb + bS) * IH + ix];
      return 0.f;
    } else {
      return in[((size_t)k * Bb + bS) * IH + oyS];
    }
  };

  const int Ks = (CK + 31) >> 5;
  v8f acc0 = {0.f,0.f,0.f,0.f,0.f,0.f,0.f,0.f};
  v8f acc1 = {0.f,0.f,0.f,0.f,0.f,0.f,0.f,0.f};

  for (int ks = 0; ks < Ks; ++ks) {
    // cooperative B-fragment staging (2 elements -> 1 packed dword per thread)
    int k0 = ks * 32 + skoff;
    float f0 = fetchB(k0);
    float f1 = fetchB(k0 + 1);
    lds[ks & 1][tid] = (unsigned int)f2bf(f0) | ((unsigned int)f2bf(f1) << 16);
    __syncthreads();

    union { uint4 q[2]; v16bf v; } Bf;
    const uint4* bq = (const uint4*)&lds[ks & 1][lane * 8];
    Bf.q[0] = bq[0];
    Bf.q[1] = bq[1];

    if (mv0) {
      union { uint4 q[2]; v16bf v; } A;
      size_t ab = ((size_t)((size_t)mt0 * Ks + ks) * 32 + lane) * 2;
      A.q[0] = pA[ab];
      A.q[1] = pA[ab + 1];
      acc0 = __builtin_amdgcn_wmma_f32_16x16x32_bf16(false, A.v, false, Bf.v,
                                                     (short)0, acc0, false, false);
    }
    if (mv1) {
      union { uint4 q[2]; v16bf v; } A;
      size_t ab = ((size_t)((size_t)mt1 * Ks + ks) * 32 + lane) * 2;
      A.q[0] = pA[ab];
      A.q[1] = pA[ab + 1];
      acc1 = __builtin_amdgcn_wmma_f32_16x16x32_bf16(false, A.v, false, Bf.v,
                                                     (short)0, acc1, false, false);
    }
  }

  // C/D layout: lane<16 -> M=mt*16+g, N=lane ; lane>=16 -> M=mt*16+8+g, N=lane-16
  const int colL = lane & 15;
  const int pcol = pt * 16 + colL;
  const int mh   = (lane < 16) ? 0 : 8;
  if (pcol < P) {
    #pragma unroll
    for (int t = 0; t < 2; ++t) {
      if (t == 0 ? !mv0 : !mv1) continue;
      const v8f& acc = (t == 0) ? acc0 : acc1;
      const int m0 = (t == 0 ? mt0 : mt1) * 16 + mh;
      #pragma unroll
      for (int g = 0; g < 8; ++g) {
        int m = m0 + g;
        if (m < M) {
          float v = acc[g];
          if (ep == 1)      { v = v * scale[m] + bias[m]; v = (v > 0.f) ? v : 0.2f * v; }
          else if (ep == 2) { v = v + bias[m]; }
          out[(size_t)m * P + pcol] = v;
        }
      }
    }
  }
}

// Pack fp32 weights [M][CK] into bf16 A-fragment order: [(mt*Ks+ks)*32+lane]*16+e
__global__ void k_pack_w(const float* __restrict__ w, unsigned short* __restrict__ pk,
                         int M, int CK)
{
  int Mt = (M + 15) >> 4, Ks = (CK + 31) >> 5;
  long long id = (long long)blockIdx.x * blockDim.x + threadIdx.x;
  long long total = (long long)Mt * Ks * 512;
  if (id >= total) return;
  int e    = (int)(id & 15);
  int lane = (int)((id >> 4) & 31);
  long long rest = id >> 9;
  int ks = (int)(rest % Ks);
  int mt = (int)(rest / Ks);
  int m  = mt * 16 + (lane & 15);
  int kl = e + ((lane >= 16) ? 8 : 0) + ((e >= 8) ? 8 : 0);
  int k  = ks * 32 + kl;
  float v = (m < M && k < CK) ? w[(size_t)m * CK + k] : 0.f;
  pk[id] = f2bf(v);
}

// pd[b,n,m] = -||x[b,n]-x[b,m]||^2 ; one block per (b,n), query row in LDS
__global__ void k_dist(const float* __restrict__ x, float* __restrict__ pd,
                       int Bb, int N, int D)
{
  __shared__ float q[256];
  int bid = blockIdx.x;              // b*N + n
  int b = bid / N, n = bid % N;
  int tid = threadIdx.x;
  if (tid < D) q[tid] = x[((size_t)b * N + n) * D + tid];
  __syncthreads();
  for (int m = tid; m < N; m += blockDim.x) {
    const float* xm = x + ((size_t)b * N + m) * D;
    float acc = 0.f;
    for (int d = 0; d < D; ++d) { float df = q[d] - xm[d]; acc += df * df; }
    pd[(size_t)bid * N + m] = -acc;
  }
}

// stable top-KNN (descending pd, earlier index wins ties -> matches lax.top_k)
__global__ void k_topk(const float* __restrict__ pd, int* __restrict__ idx,
                       int Bb, int N)
{
  long long id = (long long)blockIdx.x * blockDim.x + threadIdx.x;
  if (id >= (long long)Bb * N) return;
  const float* row = pd + (size_t)id * N;
  float bv[KNN]; int bi_[KNN];
  #pragma unroll
  for (int i = 0; i < KNN; ++i) { bv[i] = -3.4e38f; bi_[i] = 0; }
  for (int m = 0; m < N; ++m) {
    float v = row[m];
    if (v > bv[KNN - 1]) {
      int j = KNN - 1;
      while (j > 0 && bv[j - 1] < v) { bv[j] = bv[j - 1]; bi_[j] = bi_[j - 1]; --j; }
      bv[j] = v; bi_[j] = m;
    }
  }
  for (int i = 0; i < KNN; ++i) idx[(size_t)id * KNN + i] = bi_[i];
}

// feat [2D][B][N*KNN]: ch 0..D-1 = nb-ctr, ch D..2D-1 = ctr
__global__ void k_gf(const float* __restrict__ x, const int* __restrict__ idx,
                     float* __restrict__ feat, int Bb, int N, int D)
{
  long long id = (long long)blockIdx.x * blockDim.x + threadIdx.x;
  long long total = (long long)Bb * N * KNN;
  if (id >= total) return;
  int kk = (int)(id % KNN);
  int n  = (int)((id / KNN) % N);
  int b  = (int)(id / ((long long)KNN * N));
  int j  = idx[id];
  const float* xc = x + ((size_t)b * N + n) * D;
  const float* xj = x + ((size_t)b * N + j) * D;
  size_t NK = (size_t)N * KNN;
  size_t sp = (size_t)n * KNN + kk;
  for (int c = 0; c < D; ++c) {
    float cv = xc[c], nv = xj[c];
    feat[((size_t)c       * Bb + b) * NK + sp] = nv - cv;
    feat[((size_t)(c + D) * Bb + b) * NK + sp] = cv;
  }
}

// max over last spatial axis; optional transposed (b,n,c) copy for kNN reuse
__global__ void k_maxw(const float* __restrict__ in, float* __restrict__ out,
                       float* __restrict__ out_t, int C, int Bb, int OH, int OW)
{
  long long id = (long long)blockIdx.x * blockDim.x + threadIdx.x;
  long long total = (long long)C * Bb * OH;
  if (id >= total) return;
  int c = (int)(id / ((long long)Bb * OH));
  int r = (int)(id % ((long long)Bb * OH));
  int b = r / OH, oy = r % OH;
  const float* p = in + ((size_t)c * Bb + b) * ((size_t)OH * OW) + (size_t)oy * OW;
  float m = p[0];
  for (int ox = 1; ox < OW; ++ox) m = fmaxf(m, p[ox]);
  out[((size_t)c * Bb + b) * OH + oy] = m;
  if (out_t) out_t[((size_t)b * OH + oy) * C + c] = m;
}

// global max over N per (c,b), broadcast back over N
__global__ void k_gmax(const float* __restrict__ in, float* __restrict__ out,
                       int C, int Bb, int N)
{
  long long id = (long long)blockIdx.x * blockDim.x + threadIdx.x;
  if (id >= (long long)C * Bb) return;
  const float* p = in + (size_t)id * N;
  float m = p[0];
  for (int n = 1; n < N; ++n) m = fmaxf(m, p[n]);
  float* o = out + (size_t)id * N;
  for (int n = 0; n < N; ++n) o[n] = m;
}

// x'[b,n,e] = sum_d x[b,n,d] * t[b,d,e] ; t9 layout [9][B]
__global__ void k_tapply(const float* __restrict__ x, const float* __restrict__ t9,
                         float* __restrict__ xp, int Bb, int N)
{
  long long id = (long long)blockIdx.x * blockDim.x + threadIdx.x;
  if (id >= (long long)Bb * N) return;
  int b = (int)(id / N);
  float t[9];
  #pragma unroll
  for (int r = 0; r < 9; ++r) t[r] = t9[(size_t)r * Bb + b];
  const float* xi = x + (size_t)id * 3;
  float* xo = xp + (size_t)id * 3;
  #pragma unroll
  for (int e = 0; e < 3; ++e)
    xo[e] = xi[0] * t[e] + xi[1] * t[3 + e] + xi[2] * t[6 + e];
}

// [C][B][N] -> (B,N,C)
__global__ void k_transpose(const float* __restrict__ in, float* __restrict__ out,
                            int C, int Bb, int N)
{
  long long id = (long long)blockIdx.x * blockDim.x + threadIdx.x;
  if (id >= (long long)Bb * N * C) return;
  int c = (int)(id % C);
  int n = (int)((id / C) % N);
  int b = (int)(id / ((long long)C * N));
  out[id] = in[((size_t)c * Bb + b) * N + n];
}

// ---------------------------------------------------------------------------
// Workspace layout (bytes, all 256-aligned)
// ---------------------------------------------------------------------------
static constexpr size_t OFF_PD   = 0;                          // 2*1024*1024*4
static constexpr size_t OFF_IDX  = OFF_PD   + 8388608;         // 2*1024*20*4
static constexpr size_t OFF_XP   = OFF_IDX  + 163840;          // 2*1024*3*4
static constexpr size_t OFF_X1T  = OFF_XP   + 24576;           // 2*1024*256*4
static constexpr size_t OFF_X2T  = OFF_X1T  + 2097152;
static constexpr size_t OFF_FEAT = OFF_X2T  + 2097152;         // 512*2*20480*4
static constexpr size_t OFF_R1   = OFF_FEAT + 83886080;        // 256*2*20480*4
static constexpr size_t OFF_R2   = OFF_R1   + 41943040;
static constexpr size_t OFF_HBUF = OFF_R2   + 41943040;        // 1792*2*1024*4
static constexpr size_t OFF_S0   = OFF_HBUF + 14680064;        // 1024*2*4
static constexpr size_t OFF_S1   = OFF_S0   + 8192;            // 512*2*4
static constexpr size_t OFF_S2   = OFF_S1   + 4096;            // 256*2*4
static constexpr size_t OFF_S3   = OFF_S2   + 2048;            // 9*2*4
static constexpr size_t OFF_WP   = OFF_S3   + 256;             // packed weights

extern "C" void kernel_launch(void* const* d_in, const int* in_sizes, int n_in,
                              void* d_out, int out_size, void* d_ws, size_t ws_size,
                              hipStream_t stream)
{
  (void)in_sizes; (void)n_in; (void)out_size; (void)ws_size;
  const int B = 2, N = 1024;
  char* ws = (char*)d_ws;
  auto F = [&](size_t off) { return (float*)(ws + off); };
  auto IN = [&](int i) { return (const float*)d_in[i]; };

  // --- pack all conv/linear weights into WMMA A-fragment layout -------------
  struct WL { int widx; int M; int CK; };
  const WL wl[15] = {
    {1,   64,   54}, {4,  128,  576}, {7, 1024,  384}, {10, 512, 1024},
    {13, 256,  512}, {16,   9,  256}, {18, 256,   54}, {21, 256, 2304},
    {24, 256, 4608}, {27, 256, 2304}, {30, 256, 4608}, {33,1024, 2304},
    {36, 256, 5376}, {39, 256,  256}, {42, 128,  256}
  };
  size_t woff[15]; size_t cur = OFF_WP;
  for (int i = 0; i < 15; ++i) {
    woff[i] = cur;
    int Mt = (wl[i].M + 15) / 16, Ks = (wl[i].CK + 31) / 32;
    long long tot = (long long)Mt * Ks * 512;
    k_pack_w<<<(int)((tot + 255) / 256), 256, 0, stream>>>(
        IN(wl[i].widx), (unsigned short*)(ws + woff[i]), wl[i].M, wl[i].CK);
    cur += (size_t)Mt * 16 * Ks * 32 * 2;     // multiples of 256 by construction
  }

  auto gemm = [&](int mode, int wi, const float* inp, float* outp,
                  const float* sc, const float* bi, int M, int CK, int P, int ep,
                  int OH, int OW, int IH, int IW, int pad) {
    const uint4* pA = (const uint4*)(ws + woff[wi]);
    int Mt = (M + 15) / 16, Pt = (P + 15) / 16;
    int MtG = (Mt + 15) / 16;
    int blocks = MtG * Pt;
    if (mode == 0)
      k_gemm<0><<<blocks, 256, 0, stream>>>(pA, inp, outp, sc, bi, M, CK, P, ep, B, OH, OW, IH, IW, pad);
    else if (mode == 1)
      k_gemm<1><<<blocks, 256, 0, stream>>>(pA, inp, outp, sc, bi, M, CK, P, ep, B, OH, OW, IH, IW, pad);
    else
      k_gemm<2><<<blocks, 256, 0, stream>>>(pA, inp, outp, sc, bi, M, CK, P, ep, B, OH, OW, IH, IW, pad);
  };

  float* pd  = F(OFF_PD);
  int*  idx  = (int*)(ws + OFF_IDX);
  float* xp  = F(OFF_XP);
  float* x1t = F(OFF_X1T);
  float* x2t = F(OFF_X2T);
  float* feat= F(OFF_FEAT);
  float* R1  = F(OFF_R1);
  float* R2  = F(OFF_R2);
  float* hb  = F(OFF_HBUF);
  const float* x = IN(0);
  const int BN = B * N;

  auto knn_gf = [&](const float* pts, int D) {
    k_dist<<<BN, 256, 0, stream>>>(pts, pd, B, N, D);
    k_topk<<<(BN + 255) / 256, 256, 0, stream>>>(pd, idx, B, N);
    long long g = (long long)BN * KNN;
    k_gf<<<(int)((g + 255) / 256), 256, 0, stream>>>(pts, idx, feat, B, N, D);
  };

  // ---- stage 0: graph feature on raw x, transform net ----------------------
  knn_gf(x, 3);
  gemm(0, 0, feat, R1, IN(2),  IN(3),   64,   54, B*1022*18, 1, 1022, 18, 1024, 20, 0);
  gemm(0, 1, R1,   R2, IN(5),  IN(6),  128,  576, B*1020*16, 1, 1020, 16, 1022, 18, 0);
  { long long g = (long long)128*B*1020;
    k_maxw<<<(int)((g+255)/256),256,0,stream>>>(R2, R1, nullptr, 128, B, 1020, 16); }
  gemm(1, 2, R1,   R2, IN(8),  IN(9), 1024,  384, B*1018,    1, 1018, 0, 1020, 0, 0);
  { long long g = (long long)1024*B;
    k_maxw<<<(int)((g+255)/256),256,0,stream>>>(R2, F(OFF_S0), nullptr, 1024, B, 1, 1018); }
  gemm(2, 3, F(OFF_S0), F(OFF_S1), IN(11), IN(12), 512, 1024, B, 1, 1, 0, 1, 0, 0);
  gemm(2, 4, F(OFF_S1), F(OFF_S2), IN(14), IN(15), 256,  512, B, 1, 1, 0, 1, 0, 0);
  gemm(2, 5, F(OFF_S2), F(OFF_S3), nullptr, IN(17),  9,  256, B, 2, 1, 0, 1, 0, 0);
  k_tapply<<<(BN + 255) / 256, 256, 0, stream>>>(x, F(OFF_S3), xp, B, N);

  // ---- stage 1: edge conv 1/2 on transformed points ------------------------
  knn_gf(xp, 3);
  gemm(0, 6, feat, R1, IN(19), IN(20), 256,   54, B*N*20, 1, 1024, 20, 1024, 20, 1);
  gemm(0, 7, R1,   R2, IN(22), IN(23), 256, 2304, B*N*20, 1, 1024, 20, 1024, 20, 1);
  { long long g = (long long)256*BN;
    k_maxw<<<(int)((g+255)/256),256,0,stream>>>(R2, hb + (size_t)1024*BN, x1t, 256, B, N, 20); }

  // ---- stage 2: edge conv 3/4 on x1 ----------------------------------------
  knn_gf(x1t, 256);
  gemm(0, 8, feat, R1, IN(25), IN(26), 256, 4608, B*N*20, 1, 1024, 20, 1024, 20, 1);
  gemm(0, 9, R1,   R2, IN(28), IN(29), 256, 2304, B*N*20, 1, 1024, 20, 1024, 20, 1);
  { long long g = (long long)256*BN;
    k_maxw<<<(int)((g+255)/256),256,0,stream>>>(R2, hb + (size_t)1280*BN, x2t, 256, B, N, 20); }

  // ---- stage 3: edge conv 5 on x2 ------------------------------------------
  knn_gf(x2t, 256);
  gemm(0, 10, feat, R1, IN(31), IN(32), 256, 4608, B*N*20, 1, 1024, 20, 1024, 20, 1);
  { long long g = (long long)256*BN;
    k_maxw<<<(int)((g+255)/256),256,0,stream>>>(R1, hb + (size_t)1536*BN, nullptr, 256, B, N, 20); }

  // ---- conv6 on cat(x1,x2,x3), global max, conv8..10 -----------------------
  gemm(1, 11, hb + (size_t)1024*BN, R2, IN(34), IN(35), 1024, 2304, BN, 1, 1024, 0, 1024, 0, 1);
  { long long g = (long long)1024*B;
    k_gmax<<<(int)((g+255)/256),256,0,stream>>>(R2, hb, 1024, B, N); }
  gemm(1, 12, hb, R1, IN(37), IN(38), 256, 5376, BN, 1, 1024, 0, 1024, 0, 1);
  gemm(2, 13, R1, R2, IN(40), IN(41), 256,  256, BN, 1, 1024, 0, 1024, 0, 0);
  gemm(2, 14, R2, R1, IN(43), IN(44), 128,  256, BN, 1, 1024, 0, 1024, 0, 0);

  { long long g = (long long)BN * 128;
    k_transpose<<<(int)((g+255)/256),256,0,stream>>>(R1, (float*)d_out, 128, B, N); }
}
